// sparse_Convolution2D_84241488544268
// MI455X (gfx1250) — matrix operand order repaired and checked by reading, not compiled, not run on hardware
//
#include <hip/hip_runtime.h>

typedef _Float16 v16h __attribute__((ext_vector_type(16)));
typedef _Float16 v8h  __attribute__((ext_vector_type(8)));
typedef float    v8f  __attribute__((ext_vector_type(8)));

#define DIL     2
#define C_IN    64
#define H_IN    256
#define W_IN    256
#define NF      128
#define H_OUT   252
#define W_OUT   252
#define K_TOTAL (C_IN * 9)            // 576
#define K_CHUNK 32
#define N_CHUNKS (K_TOTAL / K_CHUNK)  // 18
#define M_TILE  128
#define N_TILE  128                   // one row covered by 2 tiles
#define W_ELEMS (NF * K_TOTAL)        // 73728
#define WF16_BYTES (W_ELEMS * 2)      // 147456

__device__ __forceinline__ unsigned pack_f16x2(float lo, float hi) {
  _Float16 l = (_Float16)lo, h = (_Float16)hi;   // RNE v_cvt_f16_f32
  unsigned short ul, uh;
  __builtin_memcpy(&ul, &l, 2);
  __builtin_memcpy(&uh, &h, 2);
  return ((unsigned)uh << 16) | (unsigned)ul;
}

// ---- kernel 0: one-shot fp32 -> f16 weight conversion into scratch ----
__global__ __launch_bounds__(256) void convert_weights(
    const float* __restrict__ Wt, _Float16* __restrict__ Wf16)
{
  int p = blockIdx.x * 256 + threadIdx.x;     // float4 index, 18432 total
  float4 w = ((const float4*)Wt)[p];
  uint2 pk;
  pk.x = pack_f16x2(w.x, w.y);
  pk.y = pack_f16x2(w.z, w.w);
  *(uint2*)&Wf16[4 * p] = pk;
}

// Stage one 32(k) x 128(n) f16 input chunk into Bbuf[n][kk].
// kk-major thread mapping: exactly one k->(c,i,j) decomposition per thread.
__device__ __forceinline__ void stage_B(_Float16* __restrict__ Bbuf,
                                        const float* __restrict__ X,
                                        int img, int y, int x0, int k0, int tid)
{
  const int kk = tid >> 3;        // 0..31
  const int nb = tid & 7;         // 0..7
  const int k  = k0 + kk;
  const int c  = k / 9;   const int rem = k - c * 9;
  const int i  = rem / 3; const int j   = rem - i * 3;
  const int rowbase = ((img * C_IN + c) * H_IN + (y + DIL * i)) * W_IN;
  const int colbase = x0 + DIL * j;
  _Float16* dst = &Bbuf[nb * K_CHUNK + kk];
  if (x0 == 0) {
    // fast path (block-uniform): col = colbase+n <= 131 < 256, no clamp.
    // loads/stores collapse to immediate offsets off two base addresses.
    const float* src = &X[rowbase + colbase + nb];
#pragma unroll
    for (int r = 0; r < 16; ++r)
      dst[(r << 3) * K_CHUNK] = (_Float16)src[r << 3];
  } else {
#pragma unroll
    for (int r = 0; r < 16; ++r) {
      int col = colbase + nb + (r << 3);           // clamp keeps tail tile in-bounds
      col = col > (W_IN - 1) ? (W_IN - 1) : col;
      dst[(r << 3) * K_CHUNK] = (_Float16)X[rowbase + col];
    }
  }
}

__global__ __launch_bounds__(256) void dconv_wmma_f16(
    const float* __restrict__ X, const float* __restrict__ Wt,
    const _Float16* __restrict__ Wf16,   // pre-converted weights or nullptr
    const float* __restrict__ Bias, float* __restrict__ Out)
{
  // Persistent f16 weights: [f=128][k=576] == W's native linear order. 144 KB.
  __shared__ __align__(16) _Float16 Alds[W_ELEMS];
  // Double-buffered input chunk: [buf][n=128][kk=32]. 2 x 8 KB.  Total LDS: 160 KB
  // -> exactly 2 workgroups per 320 KB WGP (4 waves / SIMD32).
  __shared__ __align__(32) _Float16 Blds[2][N_TILE * K_CHUNK];

  const int tid  = threadIdx.x;
  const int lane = tid & 31;
  const int wave = tid >> 5;   // 8 waves: 4 (M) x 2 (N)
  const int wm   = wave >> 1;  // 0..3 : 32-filter slab
  const int wn   = wave & 1;   // 0..1 : 64-pixel slab
  const int nl   = lane & 15;
  const int hi   = lane >> 4;

  int bid = blockIdx.x;
  const int xt  = bid & 1;        bid >>= 1;   // 2 x-tiles of 128 (covers 252)
  const int y   = bid % H_OUT;    bid /= H_OUT;
  const int img = bid;                          // 0..7
  const int x0  = xt * N_TILE;

  // ---- weight load into LDS ----
  if (Wf16) {
    // pre-converted: pure 128-bit copy, no VALU conversion (36 iters/thread)
    const uint4* src = (const uint4*)Wf16;
    uint4* dst = (uint4*)Alds;
#pragma unroll 4
    for (int p = tid; p < WF16_BYTES / 16; p += 256)
      dst[p] = src[p];
  } else {
    // fallback: inline fp32 -> f16 convert (72 iters/thread)
    const float4* W4 = (const float4*)Wt;
#pragma unroll 4
    for (int p = tid; p < W_ELEMS / 4; p += 256) {
      float4 w = W4[p];
      uint2 pk;
      pk.x = pack_f16x2(w.x, w.y);
      pk.y = pack_f16x2(w.z, w.w);
      *(uint2*)&Alds[4 * p] = pk;
    }
  }
  // Prologue: stage chunk 0
  stage_B(Blds[0], X, img, y, x0, 0, tid);
  __syncthreads();

  v8f acc[2][4] = {};   // [tm][tn] : wave tile = 32 filters x 64 pixels

  for (int ch = 0; ch < N_CHUNKS; ++ch) {
    const int k0 = ch * K_CHUNK;

    // Stage next chunk into the other buffer (global loads issue before WMMAs)
    if (ch + 1 < N_CHUNKS)
      stage_B(Blds[(ch + 1) & 1], X, img, y, x0, k0 + K_CHUNK, tid);

    const _Float16* Bbuf = Blds[ch & 1];

    // B fragments: b[h] = B[hi*16+h][n] -> 32 contiguous bytes in [n][kk]
    v16h bfrag[4];
#pragma unroll
    for (int tn = 0; tn < 4; ++tn) {
      int n = wn * 64 + tn * 16 + nl;
      bfrag[tn] = *(const v16h*)&Bbuf[n * K_CHUNK + hi * 16];
    }
    // A fragments from persistent LDS: K = {hi*16..+15}
#pragma unroll
    for (int tm = 0; tm < 2; ++tm) {
      int m = wm * 32 + tm * 16 + nl;
      const _Float16* arow = &Alds[m * K_TOTAL + k0];
      v8h lo = *(const v8h*)&arow[hi * 16];
      v8h up = *(const v8h*)&arow[hi * 16 + 8];
      v16h afrag = __builtin_shufflevector(lo, up,
          0, 1, 2, 3, 4, 5, 6, 7, 8, 9, 10, 11, 12, 13, 14, 15);
#pragma unroll
      for (int tn = 0; tn < 4; ++tn)
        acc[tm][tn] = __builtin_amdgcn_wmma_f32_16x16x32_f16(
            false, afrag, false, bfrag[tn],
            (short)0, acc[tm][tn], false, false);
    }

    __syncthreads();   // staging of ch+1 complete; compute of ch complete
  }

  // ---- epilogue: bias + ReLU + store (D layout: VGPR r -> M = hi*8+r, N = nl)
  float bias_reg[2][8];
#pragma unroll
  for (int tm = 0; tm < 2; ++tm)
#pragma unroll
    for (int r = 0; r < 8; ++r)
      bias_reg[tm][r] = Bias[wm * 32 + tm * 16 + hi * 8 + r];

#pragma unroll
  for (int tm = 0; tm < 2; ++tm) {
#pragma unroll
    for (int tn = 0; tn < 4; ++tn) {
      int x = x0 + wn * 64 + tn * 16 + nl;
      if (x < W_OUT) {
        int fbase = wm * 32 + tm * 16 + hi * 8;
#pragma unroll
        for (int r = 0; r < 8; ++r) {
          int f = fbase + r;
          float v = acc[tm][tn][r] + bias_reg[tm][r];
          v = v > 0.0f ? v : 0.0f;
          Out[((img * NF + f) * H_OUT + y) * W_OUT + x] = v;
        }
      }
    }
  }
}

extern "C" void kernel_launch(void* const* d_in, const int* in_sizes, int n_in,
                              void* d_out, int out_size, void* d_ws, size_t ws_size,
                              hipStream_t stream) {
  (void)in_sizes; (void)n_in; (void)out_size;
  const float* X    = (const float*)d_in[0];
  const float* Wt   = (const float*)d_in[1];
  const float* Bias = (const float*)d_in[2];
  float* Out        = (float*)d_out;

  // Pre-convert weights to f16 in scratch when it fits (deterministic per launch).
  const bool preconv = (ws_size >= (size_t)WF16_BYTES);
  _Float16* Wf16 = preconv ? (_Float16*)d_ws : nullptr;
  if (preconv) {
    // 18432 float4 / 256 threads = 72 blocks
    convert_weights<<<dim3(W_ELEMS / 4 / 256), dim3(256), 0, stream>>>(Wt, Wf16);
  }

  // 8 images * 252 rows * 2 x-tiles = 4032 workgroups of 256 threads (8 waves)
  dim3 grid(8 * H_OUT * 2);
  dim3 block(256);
  dconv_wmma_f16<<<grid, block, 0, stream>>>(X, Wt, Wf16, Bias, Out);
}
